// GraphPrefixLLMCaptioner_7249904796143
// MI455X (gfx1250) — compile-verified, matrix operand order checked by reading, not run on hardware
//
#include <hip/hip_runtime.h>
#include <hip/hip_bf16.h>
#include <math.h>

// ---------------------------------------------------------------------------
// Shapes
// ---------------------------------------------------------------------------
// B=32, T=128, R=16, A=4, C=6, D=256, H=256, V=32000
constexpr int kB = 32, kT = 128, kR = 16, kA = 4, kC = 6, kD = 256, kH = 256, kV = 32000;

typedef _Float16 v16h __attribute__((ext_vector_type(16)));
typedef float    v8f  __attribute__((ext_vector_type(8)));

// ---------------------------------------------------------------------------
// Workspace layout (bytes, all offsets multiple of 32 for v16h loads)
// ---------------------------------------------------------------------------
constexpr size_t OFF_WOUT = 0;                                  // [2000][8][512] f16
constexpr size_t OFF_WIH  = OFF_WOUT + 2000ull * 8 * 512 * 2;   // [48][24][512] f16
constexpr size_t OFF_WHH  = OFF_WIH  + 48ull * 24 * 512 * 2;    // [48][8][512]  f16
constexpr size_t OFF_WHG  = OFF_WHH  + 48ull * 8  * 512 * 2;    // [16][8][512]  f16
constexpr size_t OFF_RP   = OFF_WHG  + 16ull * 8  * 512 * 2;    // [B][R][D] f32
constexpr size_t OFF_CP   = OFF_RP   + (size_t)kB * kR * kD * 4;// [B][C][D] f32
constexpr size_t OFF_CREF = OFF_CP   + (size_t)kB * kC * kD * 4;// [B][C][D] f32
constexpr size_t OFF_H0   = OFF_CREF + (size_t)kB * kC * kD * 4;// [B][H]    f32
constexpr size_t OFF_HOUT = OFF_H0   + (size_t)kB * kH * 4;     // [T*B][H]  f16 (row M = t*32+b)

// ---------------------------------------------------------------------------
// WMMA helpers (v_wmma_f32_16x16x32_f16; wave32)
// A-frag (16x32 f16): lane l holds row (l&15); element e -> K = 8*(l>>4) + e + (e>=8 ? 8 : 0)
// B-frag: pre-swizzled so each lane loads 16 contiguous halves (32B).
// C/D-frag (16x16 f32): lane l, reg j -> (m = j + 8*(l>>4), n = l&15)
// ---------------------------------------------------------------------------
__device__ __forceinline__ v8f vzero8() {
  v8f z;
#pragma unroll
  for (int j = 0; j < 8; ++j) z[j] = 0.0f;
  return z;
}

__device__ __forceinline__ v8f wmma16(v16h a, v16h b, v8f c) {
  return __builtin_amdgcn_wmma_f32_16x16x32_f16(false, a, false, b, (short)0, c, false, false);
}

// p = base + row*stride + kt*32 + 8*(lane>>4)  -> two contiguous 8-half runs
__device__ __forceinline__ v16h frag_a(const _Float16* __restrict__ p) {
  v16h a;
#pragma unroll
  for (int e = 0; e < 8; ++e) a[e] = p[e];
#pragma unroll
  for (int e = 8; e < 16; ++e) a[e] = p[e + 8];
  return a;
}

__device__ __forceinline__ v16h frag_b(const _Float16* __restrict__ sw, int tile, int lane) {
  return *(const v16h*)(sw + (((size_t)tile) << 9) + (lane << 4));
}

__device__ __forceinline__ float sigmoidf_(float x) { return 1.0f / (1.0f + expf(-x)); }
__device__ __forceinline__ float geluf_(float x) {
  return 0.5f * x * (1.0f + erff(x * 0.70710678118654752f));  // exact erf gelu
}

// ---------------------------------------------------------------------------
// Kernel 0: convert f32 weight W[K][N] (row-major) -> f16 WMMA B-fragment
// layout out[N/16][K/32][32][16].
// ---------------------------------------------------------------------------
__global__ void swizzle_b_kernel(const float* __restrict__ W, int K, int N,
                                 _Float16* __restrict__ out) {
  const int KT = K >> 5;
  const size_t total = (size_t)(N >> 4) * KT * 512;
  for (size_t i = (size_t)blockIdx.x * blockDim.x + threadIdx.x; i < total;
       i += (size_t)gridDim.x * blockDim.x) {
    int e    = (int)(i & 15);
    int lane = (int)((i >> 4) & 31);
    size_t tile = i >> 9;
    int kt = (int)(tile % KT);
    int nt = (int)(tile / KT);
    int k = kt * 32 + ((lane >> 4) << 3) + e + (e >= 8 ? 8 : 0);
    int n = nt * 16 + (lane & 15);
    out[i] = (_Float16)W[(size_t)k * N + n];
  }
}

// ---------------------------------------------------------------------------
// Kernel 1: graph-reasoning phase, one workgroup per batch element (tiny).
// Produces r_p, c_p, refined c, h0 in workspace.
// ---------------------------------------------------------------------------
__global__ __launch_bounds__(256) void graph_kernel(
    const float* __restrict__ rfg, const float* __restrict__ anat,
    const float* __restrict__ conc, const float* __restrict__ rprior,
    const float* __restrict__ cprior, const float* __restrict__ Wr,
    const float* __restrict__ br, const float* __restrict__ Wa,
    const float* __restrict__ ba, const float* __restrict__ Wc,
    const float* __restrict__ bc, const float* __restrict__ Wg,
    const float* __restrict__ bg, const float* __restrict__ Wi,
    const float* __restrict__ bi, float* __restrict__ rp_out,
    float* __restrict__ cp_out, float* __restrict__ cref_out,
    float* __restrict__ h0_out) {
  __shared__ float rf[16][256];
  __shared__ float rp[16][256];
  __shared__ float abuf[4][256];
  __shared__ float t4[4][256];
  __shared__ float amsg[4][256];
  __shared__ float cbuf[6][256];
  __shared__ float t6[6][256];
  __shared__ float cmsg[6][256];
  __shared__ float ra[16][4];
  __shared__ float ac[4][6];
  __shared__ float meanv[256];

  const int tid = threadIdx.x;
  const int b = blockIdx.x;
  const float scale = 0.0625f;  // 1/sqrt(256)

  for (int i = tid; i < 4096; i += 256) rf[i >> 8][i & 255] = rfg[(size_t)b * 4096 + i];
  for (int i = tid; i < 1024; i += 256) abuf[i >> 8][i & 255] = anat[i];
  for (int i = tid; i < 1536; i += 256) cbuf[i / 256][i % 256] = conc[i];
  __syncthreads();

  // r_p = rf @ Wr + br
  for (int idx = tid; idx < 4096; idx += 256) {
    int r = idx >> 8, d = idx & 255;
    float acc = br[d];
    for (int k = 0; k < 256; ++k) acc += rf[r][k] * Wr[k * 256 + d];
    rp[r][d] = acc;
  }
  // aW = a @ Wa + ba
  for (int idx = tid; idx < 1024; idx += 256) {
    int j = idx >> 8, d = idx & 255;
    float acc = ba[d];
    for (int k = 0; k < 256; ++k) acc += abuf[j][k] * Wa[k * 256 + d];
    t4[j][d] = acc;
  }
  __syncthreads();

  // ra = softmax_a(scale * r_p . aW + log prior)
  if (tid < 64) {
    int r = tid >> 2, j = tid & 3;
    float s = 0.0f;
    for (int d = 0; d < 256; ++d) s += rp[r][d] * t4[j][d];
    ra[r][j] = s * scale + logf(fmaxf(rprior[r * 4 + j], 1e-8f));
  }
  __syncthreads();
  if (tid < 16) {
    float mx = ra[tid][0];
    for (int j = 1; j < 4; ++j) mx = fmaxf(mx, ra[tid][j]);
    float s = 0.0f;
    for (int j = 0; j < 4; ++j) { float e = expf(ra[tid][j] - mx); ra[tid][j] = e; s += e; }
    for (int j = 0; j < 4; ++j) ra[tid][j] /= s;
  }
  __syncthreads();

  // a = gelu(a + ((ra^T @ rf)/R) @ Wg + bg)
  for (int idx = tid; idx < 1024; idx += 256) {
    int j = idx >> 8, d = idx & 255;
    float s = 0.0f;
    for (int r = 0; r < 16; ++r) s += ra[r][j] * rf[r][d];
    amsg[j][d] = s * (1.0f / 16.0f);
  }
  __syncthreads();
  for (int idx = tid; idx < 1024; idx += 256) {
    int j = idx >> 8, d = idx & 255;
    float acc = bg[d];
    for (int k = 0; k < 256; ++k) acc += amsg[j][k] * Wg[k * 256 + d];
    abuf[j][d] = geluf_(abuf[j][d] + acc);
  }
  __syncthreads();

  // aW2, cW
  for (int idx = tid; idx < 1024; idx += 256) {
    int j = idx >> 8, d = idx & 255;
    float acc = ba[d];
    for (int k = 0; k < 256; ++k) acc += abuf[j][k] * Wa[k * 256 + d];
    t4[j][d] = acc;
  }
  for (int idx = tid; idx < 1536; idx += 256) {
    int ci = idx / 256, d = idx % 256;
    float acc = bc[d];
    for (int k = 0; k < 256; ++k) acc += cbuf[ci][k] * Wc[k * 256 + d];
    t6[ci][d] = acc;
  }
  __syncthreads();

  // ac = softmax_c(scale * aW2 . cW + log prior)
  if (tid < 24) {
    int j = tid / 6, ci = tid % 6;
    float s = 0.0f;
    for (int d = 0; d < 256; ++d) s += t4[j][d] * t6[ci][d];
    ac[j][ci] = s * scale + logf(fmaxf(cprior[j * 6 + ci], 1e-8f));
  }
  __syncthreads();
  if (tid < 4) {
    float mx = ac[tid][0];
    for (int i = 1; i < 6; ++i) mx = fmaxf(mx, ac[tid][i]);
    float s = 0.0f;
    for (int i = 0; i < 6; ++i) { float e = expf(ac[tid][i] - mx); ac[tid][i] = e; s += e; }
    for (int i = 0; i < 6; ++i) ac[tid][i] /= s;
  }
  __syncthreads();

  // c = gelu(c + ((ac^T @ a)/A) @ Wg + bg)
  for (int idx = tid; idx < 1536; idx += 256) {
    int ci = idx / 256, d = idx % 256;
    float s = 0.0f;
    for (int j = 0; j < 4; ++j) s += ac[j][ci] * abuf[j][d];
    cmsg[ci][d] = s * 0.25f;
  }
  __syncthreads();
  for (int idx = tid; idx < 1536; idx += 256) {
    int ci = idx / 256, d = idx % 256;
    float acc = bg[d];
    for (int k = 0; k < 256; ++k) acc += cmsg[ci][k] * Wg[k * 256 + d];
    cbuf[ci][d] = geluf_(cbuf[ci][d] + acc);
  }
  __syncthreads();

  // outputs: c_p = c @ Wc + bc, refined c, h0, r_p
  for (int idx = tid; idx < 1536; idx += 256) {
    int ci = idx / 256, d = idx % 256;
    float acc = bc[d];
    for (int k = 0; k < 256; ++k) acc += cbuf[ci][k] * Wc[k * 256 + d];
    cp_out[(size_t)b * 1536 + idx] = acc;
    cref_out[(size_t)b * 1536 + idx] = cbuf[ci][d];
  }
  {
    float s = 0.0f;
    for (int r = 0; r < 16; ++r) s += rf[r][tid];
    meanv[tid] = s * (1.0f / 16.0f);
  }
  __syncthreads();
  {
    float acc = bi[tid];
    for (int k = 0; k < 256; ++k) acc += meanv[k] * Wi[k * 256 + tid];
    h0_out[(size_t)b * 256 + tid] = acc;
  }
  for (int i = tid; i < 4096; i += 256) rp_out[(size_t)b * 4096 + i] = rp[i >> 8][i & 255];
}

// ---------------------------------------------------------------------------
// Kernel 2: sequential GRU decode. One persistent 1024-thread workgroup
// (32 waves) runs all T steps; WMMA for q/gates; epilogue-fused GRU update.
// All WMMA k-loops stay ROLLED (unroll(disable)) so only one A-frag / B-frag
// pair is live per wave -> no VGPR spill-to-scratch; latency hidden by the
// 32 resident waves.
// Dynamic LDS layout (bytes):
//   h      f32[32][256]    @ 0       (32768)
//   q      f32[32][256]    @ 32768   (32768)
//   rbuf   f32[32][256]    @ 65536   (32768)
//   zbuf   f32[32][256]    @ 98304   (32768)
//   attl   f32[32][16]     @ 131072  (2048)
//   tcv    f32[32][8]      @ 133120  (1024)
//   hf16   f16[32][272]    @ 134144  (17408)
//   inpf16 f16[32][776]    @ 151552  (49664)   total 201216
// ---------------------------------------------------------------------------
constexpr size_t kDecodeLds = 201216;

__global__ __launch_bounds__(1024) void decode_kernel(
    const float* __restrict__ rfg,           // [B,R,D] region_features
    const int* __restrict__ tokens,          // [B,T]
    const float* __restrict__ token_embed,   // [V,D]
    const float* __restrict__ bhg, const float* __restrict__ bih,
    const float* __restrict__ bhh,
    const _Float16* __restrict__ whg_sw,     // [16][8][512]
    const _Float16* __restrict__ wih_sw,     // [48][24][512]
    const _Float16* __restrict__ whh_sw,     // [48][8][512]
    const float* __restrict__ rp,            // [B,R,D]
    const float* __restrict__ cp,            // [B,C,D]
    const float* __restrict__ cref,          // [B,C,D]
    const float* __restrict__ h0,            // [B,H]
    _Float16* __restrict__ hout)             // [T*32][256], row M = t*32+b
{
  extern __shared__ char smem[];
  float*    h      = (float*)(smem);
  float*    q      = (float*)(smem + 32768);
  float*    rbuf   = (float*)(smem + 65536);
  float*    zbuf   = (float*)(smem + 98304);
  float*    attl   = (float*)(smem + 131072);
  float*    tcv    = (float*)(smem + 133120);
  _Float16* hf16   = (_Float16*)(smem + 134144);
  _Float16* inpf16 = (_Float16*)(smem + 151552);

  const int tid  = threadIdx.x;
  const int lane = tid & 31;
  const int w    = tid >> 5;
  const int mt   = w & 1;        // 2 M-tiles (B=32)
  const int nt   = w >> 1;       // 16 N-tiles (H=256)
  const int half8 = (lane >> 4) << 3;
  const int n16   = lane & 15;
  const int arow  = mt * 16 + n16;   // A-fragment row for this lane

  for (int i = tid; i < 8192; i += 1024) h[i] = h0[i];
  __syncthreads();

#pragma clang loop unroll(disable)
  for (int t = 0; t < kT; ++t) {
    // -- snapshot h -> f16 (WMMA A source) --
    for (int i = tid; i < 8192; i += 1024) {
      int bb = i >> 8, d = i & 255;
      hf16[bb * 272 + d] = (_Float16)h[i];
    }
    __syncthreads();

    // -- q = h @ Whg + bhg : M=32,N=256,K=256 -> one 16x16 tile per wave --
    {
      v8f acc = vzero8();
#pragma clang loop unroll(disable)
      for (int kt = 0; kt < 8; ++kt) {
        v16h a = frag_a(hf16 + arow * 272 + kt * 32 + half8);
        v16h b = frag_b(whg_sw, nt * 8 + kt, lane);
        acc = wmma16(a, b, acc);
      }
#pragma unroll
      for (int j = 0; j < 8; ++j) {
        int m = j + half8;
        int row = mt * 16 + m;
        int col = nt * 16 + n16;
        q[row * 256 + col] = acc[j] + bhg[col];
      }
    }
    __syncthreads();

    // -- attention logits / token-concept scores --
    if (tid < 512) {
      int bb = tid >> 4, r = tid & 15;
      const float* rpp = rp + (((size_t)bb * 16 + r) << 8);
      const float* qq = q + (bb << 8);
      float s = 0.0f;
      for (int k = 0; k < 256; ++k) s += rpp[k] * qq[k];
      attl[bb * 16 + r] = s * 0.0625f;
    } else if (tid < 704) {
      int i2 = tid - 512;
      int bb = i2 / 6, ci = i2 % 6;
      const float* cpp = cp + (((size_t)bb * 6 + ci) << 8);
      const float* qq = q + (bb << 8);
      float s = 0.0f;
      for (int k = 0; k < 256; ++k) s += cpp[k] * qq[k];
      tcv[bb * 8 + ci] = sigmoidf_(s);
    }
    __syncthreads();
    if (tid < 32) {          // softmax over regions
      int bb = tid;
      float mx = attl[bb * 16];
      for (int r = 1; r < 16; ++r) mx = fmaxf(mx, attl[bb * 16 + r]);
      float s = 0.0f;
      for (int r = 0; r < 16; ++r) { float e = expf(attl[bb * 16 + r] - mx); attl[bb * 16 + r] = e; s += e; }
      for (int r = 0; r < 16; ++r) attl[bb * 16 + r] /= s;
    } else if (tid < 64) {   // normalize sigmoid scores
      int bb = tid - 32;
      float s = 0.0f;
      for (int ci = 0; ci < 6; ++ci) s += tcv[bb * 8 + ci];
      s = fmaxf(s, 1e-8f);
      for (int ci = 0; ci < 6; ++ci) tcv[bb * 8 + ci] /= s;
    }
    __syncthreads();

    // -- build GRU input [x_emb | region_ctx | concept_ctx] directly as f16 --
    for (int i = tid; i < 24576; i += 1024) {
      int bb = i / 768, col = i - bb * 768;
      float v;
      if (col < 256) {
        int tk = tokens[bb * kT + t];
        v = token_embed[(size_t)tk * 256 + col];
      } else if (col < 512) {
        int d = col - 256;
        float s = 0.0f;
        for (int r = 0; r < 16; ++r) s += attl[bb * 16 + r] * rfg[(((size_t)bb * 16 + r) << 8) + d];
        v = s;
      } else {
        int d = col - 512;
        float s = 0.0f;
        for (int ci = 0; ci < 6; ++ci) s += tcv[bb * 8 + ci] * cref[(((size_t)bb * 6 + ci) << 8) + d];
        v = s;
      }
      inpf16[bb * 776 + col] = (_Float16)v;
    }
    __syncthreads();

    // -- r and z gates: sigmoid(inp@Wih + h@Whh + biases), cols [0,512) --
#pragma clang loop unroll(disable)
    for (int g = 0; g < 2; ++g) {
      int ntg = g * 16 + nt;
      v8f acc = vzero8();
#pragma clang loop unroll(disable)
      for (int kt = 0; kt < 24; ++kt) {
        v16h a = frag_a(inpf16 + arow * 776 + kt * 32 + half8);
        v16h b = frag_b(wih_sw, ntg * 24 + kt, lane);
        acc = wmma16(a, b, acc);
      }
#pragma clang loop unroll(disable)
      for (int kt = 0; kt < 8; ++kt) {
        v16h a = frag_a(hf16 + arow * 272 + kt * 32 + half8);
        v16h b = frag_b(whh_sw, ntg * 8 + kt, lane);
        acc = wmma16(a, b, acc);
      }
      float* dst = g ? zbuf : rbuf;
#pragma unroll
      for (int j = 0; j < 8; ++j) {
        int m = j + half8;
        int row = mt * 16 + m;
        int col = nt * 16 + n16;
        int gc = ntg * 16 + n16;
        dst[row * 256 + col] = sigmoidf_(acc[j] + bih[gc] + bhh[gc]);
      }
    }
    __syncthreads();

    // -- n gate (cols [512,768)) kept as two accs; fused GRU hidden update --
    {
      int ntg = 32 + nt;
      v8f acc1 = vzero8();  // inp @ Wih[:,512:768]
      v8f acc2 = vzero8();  // h   @ Whh[:,512:768]
#pragma clang loop unroll(disable)
      for (int kt = 0; kt < 24; ++kt) {
        v16h a = frag_a(inpf16 + arow * 776 + kt * 32 + half8);
        v16h b = frag_b(wih_sw, ntg * 24 + kt, lane);
        acc1 = wmma16(a, b, acc1);
      }
#pragma clang loop unroll(disable)
      for (int kt = 0; kt < 8; ++kt) {
        v16h a = frag_a(hf16 + arow * 272 + kt * 32 + half8);
        v16h b = frag_b(whh_sw, ntg * 8 + kt, lane);
        acc2 = wmma16(a, b, acc2);
      }
#pragma unroll
      for (int j = 0; j < 8; ++j) {
        int m = j + half8;
        int row = mt * 16 + m;           // batch index b
        int col = nt * 16 + n16;         // hidden index
        int gc = 512 + col;
        float rr = rbuf[row * 256 + col];
        float zz = zbuf[row * 256 + col];
        float nn = tanhf(acc1[j] + bih[gc] + rr * (acc2[j] + bhh[gc]));
        float hnew = (1.0f - zz) * nn + zz * h[row * 256 + col];
        h[row * 256 + col] = hnew;
        hout[(((size_t)t * 32 + row) << 8) + col] = (_Float16)hnew;
      }
    }
    __syncthreads();
  }
}

// ---------------------------------------------------------------------------
// Kernel 3: logits = H_all[4096,256] @ Wout[256,32000] + bout (WMMA f16).
// Block tile: M=128 x N=64, 8 waves; each wave: 1 M-tile x 4 N-tiles.
// Wout (16MB f16 swizzled) stays L2-resident across the grid. k-loop rolled
// so only 4 B-frags + 1 A-frag are live (no spills).
// ---------------------------------------------------------------------------
constexpr size_t kLogitsLds = 128 * 264 * 2;  // 67584 B

__global__ __launch_bounds__(256) void logits_kernel(
    const _Float16* __restrict__ hout, const _Float16* __restrict__ wout_sw,
    const float* __restrict__ bout, float* __restrict__ out) {
  extern __shared__ char smem[];
  _Float16* at = (_Float16*)smem;  // [128][264]

  const int tid = threadIdx.x, lane = tid & 31, w = tid >> 5;
  const int mrow0 = blockIdx.y * 128;
  const int nt0 = blockIdx.x * 4;

  // stage A tile as b128 chunks (padded stride 264 halves keeps 16B align)
  for (int i = tid; i < 4096; i += 256) {
    int r = i >> 5, c8 = (i & 31) << 3;
    *(uint4*)(at + r * 264 + c8) =
        *(const uint4*)(hout + (((size_t)(mrow0 + r)) << 8) + c8);
  }
  __syncthreads();

  const int half8 = (lane >> 4) << 3, n16 = lane & 15;
  const int arow = w * 16 + n16;

  v8f acc[4];
#pragma unroll
  for (int ni = 0; ni < 4; ++ni) acc[ni] = vzero8();

#pragma clang loop unroll(disable)
  for (int kt = 0; kt < 8; ++kt) {
    v16h a = frag_a(at + arow * 264 + kt * 32 + half8);
#pragma unroll
    for (int ni = 0; ni < 4; ++ni) {
      v16h b = frag_b(wout_sw, (nt0 + ni) * 8 + kt, lane);
      acc[ni] = wmma16(a, b, acc[ni]);
    }
  }

#pragma unroll
  for (int ni = 0; ni < 4; ++ni) {
    int n = (nt0 + ni) * 16 + n16;
    float bo = bout[n];
#pragma unroll
    for (int j = 0; j < 8; ++j) {
      int m = j + half8;
      int M = mrow0 + w * 16 + m;  // M = t*32 + b
      int bb = M & 31, tt = M >> 5;
      out[((size_t)(bb * kT + tt)) * kV + n] = acc[ni][j] + bo;
    }
  }
}

// ---------------------------------------------------------------------------
// Launch
// ---------------------------------------------------------------------------
extern "C" void kernel_launch(void* const* d_in, const int* in_sizes, int n_in,
                              void* d_out, int out_size, void* d_ws, size_t ws_size,
                              hipStream_t stream) {
  (void)in_sizes; (void)n_in; (void)out_size; (void)ws_size;
  const float* region_features = (const float*)d_in[0];
  const int*   tokens          = (const int*)d_in[1];
  const float* token_embed     = (const float*)d_in[2];
  const float* concept_embed   = (const float*)d_in[3];
  const float* anatomy_embed   = (const float*)d_in[4];
  const float* region_prior    = (const float*)d_in[5];
  const float* concept_prior   = (const float*)d_in[6];
  const float* Wr = (const float*)d_in[7];  const float* br = (const float*)d_in[8];
  const float* Wa = (const float*)d_in[9];  const float* ba = (const float*)d_in[10];
  const float* Wc = (const float*)d_in[11]; const float* bc = (const float*)d_in[12];
  const float* Wg = (const float*)d_in[13]; const float* bg = (const float*)d_in[14];
  const float* Wi = (const float*)d_in[15]; const float* bi = (const float*)d_in[16];
  const float* Whg = (const float*)d_in[17]; const float* bhg = (const float*)d_in[18];
  const float* Wih = (const float*)d_in[19]; const float* bih = (const float*)d_in[20];
  const float* Whh = (const float*)d_in[21]; const float* bhh = (const float*)d_in[22];
  const float* Wout = (const float*)d_in[23]; const float* bout = (const float*)d_in[24];

  char* ws = (char*)d_ws;
  _Float16* wout_sw = (_Float16*)(ws + OFF_WOUT);
  _Float16* wih_sw  = (_Float16*)(ws + OFF_WIH);
  _Float16* whh_sw  = (_Float16*)(ws + OFF_WHH);
  _Float16* whg_sw  = (_Float16*)(ws + OFF_WHG);
  float* rp_ws   = (float*)(ws + OFF_RP);
  float* cp_ws   = (float*)(ws + OFF_CP);
  float* cref_ws = (float*)(ws + OFF_CREF);
  float* h0_ws   = (float*)(ws + OFF_H0);
  _Float16* hout_ws = (_Float16*)(ws + OFF_HOUT);

  // Weight conversion + B-fragment swizzle (per call; deterministic).
  swizzle_b_kernel<<<2048, 256, 0, stream>>>(Wout, 256, kV, wout_sw);
  swizzle_b_kernel<<<512, 256, 0, stream>>>(Wih, 768, 768, wih_sw);
  swizzle_b_kernel<<<256, 256, 0, stream>>>(Whh, 256, 768, whh_sw);
  swizzle_b_kernel<<<64, 256, 0, stream>>>(Whg, 256, 256, whg_sw);

  // Graph reasoning phase (tiny): one workgroup per batch element.
  graph_kernel<<<kB, 256, 0, stream>>>(
      region_features, anatomy_embed, concept_embed, region_prior, concept_prior,
      Wr, br, Wa, ba, Wc, bc, Wg, bg, Wi, bi, rp_ws, cp_ws, cref_ws, h0_ws);

  // Sequential GRU decode: one persistent workgroup, ~197KB LDS (<=320KB/WGP).
  (void)hipFuncSetAttribute(reinterpret_cast<const void*>(decode_kernel),
                            hipFuncAttributeMaxDynamicSharedMemorySize,
                            (int)kDecodeLds);
  decode_kernel<<<1, 1024, kDecodeLds, stream>>>(
      region_features, tokens, token_embed, bhg, bih, bhh, whg_sw, wih_sw,
      whh_sw, rp_ws, cp_ws, cref_ws, h0_ws, hout_ws);

  // Vocab projection: 4096x32000x256 WMMA GEMM (dominant compute + HBM stores).
  (void)hipFuncSetAttribute(reinterpret_cast<const void*>(logits_kernel),
                            hipFuncAttributeMaxDynamicSharedMemorySize,
                            (int)kLogitsLds);
  logits_kernel<<<dim3(kV / 64, (kB * kT) / 128), 256, kLogitsLds, stream>>>(
      hout_ws, wout_sw, bout, (float*)d_out);
}
